// RNNModel_17308718203030
// MI455X (gfx1250) — matrix operand order repaired and checked by reading, not compiled, not run on hardware
//
#include <hip/hip_runtime.h>
#include <hip/hip_bf16.h>
#include <stdint.h>

// ---------------- model dims ----------------
#define NHID     1024
#define G4H      4096          // 4*NHID
#define KCAT     2048          // 2*NHID ([x;h] concat)
#define TSEQ     128
#define BATCH    64
#define NROWS    8192          // T*B
#define NPC      225
#define NCLASSES 224
#define NPC_PAD  240           // 15 * 16

// ---------------- vector types ----------------
typedef __attribute__((ext_vector_type(16))) __bf16   v16bf;
typedef __attribute__((ext_vector_type(8)))  float    v8f;
typedef __attribute__((ext_vector_type(16))) uint16_t v16u;
typedef __attribute__((ext_vector_type(8)))  uint16_t v8u;
typedef __attribute__((ext_vector_type(4)))  int      v4i;

#if defined(__AMDGCN__) && __has_builtin(__builtin_amdgcn_global_load_async_to_lds_b128)
#define USE_ASYNC 1
#else
#define USE_ASYNC 0
#endif

__device__ __forceinline__ void wait_async0() {
#if defined(__AMDGCN__)
#if __has_builtin(__builtin_amdgcn_s_wait_asynccnt)
  __builtin_amdgcn_s_wait_asynccnt(0);
#else
  asm volatile("s_wait_asynccnt 0" ::: "memory");
#endif
#endif
}

__device__ __forceinline__ uint16_t f2bf(float f) {
  uint32_t u = __float_as_uint(f);
  u += 0x7FFFu + ((u >> 16) & 1u);     // round-to-nearest-even
  return (uint16_t)(u >> 16);
}

__device__ __forceinline__ v8f zero8f() {
  v8f a;
#pragma unroll
  for (int j = 0; j < 8; ++j) a[j] = 0.f;
  return a;
}
__device__ __forceinline__ v16u zero16u() {
  v16u a;
#pragma unroll
  for (int j = 0; j < 16; ++j) a[j] = 0;
  return a;
}

__device__ __forceinline__ v8f wmma_bf16(v16u a, v16u b, v8f c) {
  return __builtin_amdgcn_wmma_f32_16x16x32_bf16(
      false, __builtin_bit_cast(v16bf, a),
      false, __builtin_bit_cast(v16bf, b),
      (short)0, c, false, false);
}

// A-tile (16x32 bf16, M x K) loader. rowptr = &A[row][k0] for THIS lane's row.
// Layout: lanes 0-15 row M=lane, a[0..15]=K0..15;
//         lanes 16-31 row M=lane-16, a[0..15]=K16..31.
__device__ __forceinline__ v16u load_a(const uint16_t* rowptr, int lane) {
  int kb = (lane < 16) ? 0 : 16;
  v8u lo = *(const v8u*)(rowptr + kb);
  v8u hi = *(const v8u*)(rowptr + kb + 8);
  v16u a;
#pragma unroll
  for (int j = 0; j < 8; ++j) { a[j] = lo[j]; a[8 + j] = hi[j]; }
  return a;
}

// B-tile (32x16, K x N) from a row-major [N,K] bf16 matrix: B[k][n] = W[n0+n][k0+k].
__device__ __forceinline__ v16u load_b_nk(const uint16_t* W, int ldk, int n0,
                                          int k0, int lane) {
  int n = n0 + (lane & 15);
  int kb = (lane < 16) ? 0 : 16;
  return *(const v16u*)(W + (size_t)n * ldk + k0 + kb);
}

__device__ __forceinline__ float sigm(float x) { return 1.f / (1.f + __expf(-x)); }
__device__ __forceinline__ float wave_max(float v) {
  for (int o = 16; o; o >>= 1) v = fmaxf(v, __shfl_xor(v, o, 32));
  return v;
}
__device__ __forceinline__ float wave_sum(float v) {
  for (int o = 16; o; o >>= 1) v += __shfl_xor(v, o, 32);
  return v;
}

// ---------------- prep kernels ----------------
__global__ void k_prep_wcat(const float* __restrict__ Wih,
                            const float* __restrict__ Whh,
                            uint16_t* __restrict__ wcat) {
  int idx = blockIdx.x * 256 + threadIdx.x;         // 2*4096*2048
  int l = idx >> 23;
  int rem = idx & ((1 << 23) - 1);
  int n = rem >> 11;
  int k = rem & 2047;
  float v = (k < NHID) ? Wih[(size_t)l * G4H * NHID + (size_t)n * NHID + k]
                       : Whh[(size_t)l * G4H * NHID + (size_t)n * NHID + (k - NHID)];
  wcat[idx] = f2bf(v);
}

__global__ void k_prep_bias(const float* __restrict__ bih,
                            const float* __restrict__ bhh,
                            float* __restrict__ bc) {
  int i = blockIdx.x * 256 + threadIdx.x;           // 2*4096
  bc[i] = bih[i] + bhh[i];
}

// topWT[c][d] bf16 = top_W[d][c]   (row-major [N=224, K=1024])
__global__ void k_prep_topwt(const float* __restrict__ topW,
                             uint16_t* __restrict__ topwt) {
  int idx = blockIdx.x * 256 + threadIdx.x;         // 224*1024
  int c = idx >> 10;
  int d = idx & 1023;
  topwt[idx] = f2bf(topW[(size_t)d * NCLASSES + c]);
}

__global__ void k_embed(const int* __restrict__ ids,
                        const float* __restrict__ embW,
                        uint16_t* __restrict__ xbf) {
  int idx = blockIdx.x * 256 + threadIdx.x;         // 8192*1024
  int row = idx >> 10;
  int col = idx & 1023;
  int id = ids[row];
  xbf[idx] = f2bf(embW[(size_t)id * NHID + col]);
}

__global__ void k_init_state(const float* __restrict__ h0l,
                             const float* __restrict__ c0l,
                             float* __restrict__ hst, float* __restrict__ cst,
                             uint16_t* __restrict__ hbf) {
  int i = blockIdx.x * 256 + threadIdx.x;           // 64*1024
  float h = h0l[i];
  hst[i] = h;
  cst[i] = c0l[i];
  hbf[i] = f2bf(h);
}

// ---------------- LSTM step: g = [x_t;h] @ Wcat^T  (M=64,N=4096,K=2048) ----------------
// 32 blocks x 256 threads (8 waves). Block stages 128 Wcat rows x 32-K chunk into
// LDS (double-buffered, async copy). Each wave: one 16-wide N-tile, all 4 M-tiles.
__global__ void k_step_gemm(const uint16_t* __restrict__ xseq,
                            const uint16_t* __restrict__ hbf,
                            const uint16_t* __restrict__ wcat,
                            float* __restrict__ gbuf, int t) {
  __shared__ __align__(128) uint16_t ldsB[2][128 * 32];   // 2 x 8KB
  const int tid = threadIdx.x;
  const int lane = tid & 31;
  const int w = tid >> 5;                 // wave 0..7 -> n-tile
  const int n0_blk = blockIdx.x * 128;

  // staging assignment: 2 threads per Wcat row, 32B each
  const int crow = tid >> 1;              // 0..127
  const int chalf = tid & 1;
  const uint16_t* gsrc_row = wcat + (size_t)(n0_blk + crow) * KCAT + chalf * 16;
  uint16_t* lrow0 = &ldsB[0][0] + crow * 32 + chalf * 16;
  uint16_t* lrow1 = &ldsB[1][0] + crow * 32 + chalf * 16;

  auto stage = [&](int k0, int buf) {
    const uint16_t* g = gsrc_row + k0;
    uint16_t* l = buf ? lrow1 : lrow0;
#if USE_ASYNC
    __builtin_amdgcn_global_load_async_to_lds_b128((v4i*)g, (v4i*)l, 0, 0);
    __builtin_amdgcn_global_load_async_to_lds_b128((v4i*)(g + 8), (v4i*)(l + 8), 0, 0);
#else
    v8u t0 = *(const v8u*)g;
    v8u t1 = *(const v8u*)(g + 8);
    *(v8u*)l = t0;
    *(v8u*)(l + 8) = t1;
#endif
  };

  // A row pointers for the 4 M-tiles
  const uint16_t* xrow[4];
  const uint16_t* hrow[4];
#pragma unroll
  for (int mi = 0; mi < 4; ++mi) {
    int m = mi * 16 + (lane & 15);
    xrow[mi] = xseq + (size_t)(t * BATCH + m) * NHID;
    hrow[mi] = hbf + (size_t)m * NHID;
  }

  v8f acc[4];
#pragma unroll
  for (int mi = 0; mi < 4; ++mi) acc[mi] = zero8f();

  // prologue: stage chunk 0
  stage(0, 0);
#if USE_ASYNC
  wait_async0();
#endif
  __syncthreads();

  const uint16_t* bbase = &ldsB[0][0] + (w * 16 + (lane & 15)) * 32 + ((lane < 16) ? 0 : 16);
  for (int i = 0; i < 64; ++i) {
    int k0 = i * 32;
    int cur = i & 1;
#if USE_ASYNC
    if (i + 1 < 64) stage((i + 1) * 32, (i + 1) & 1);
#endif
    v16u b = *(const v16u*)(bbase + cur * (128 * 32));
#pragma unroll
    for (int mi = 0; mi < 4; ++mi) {
      const uint16_t* arow = (k0 < NHID) ? (xrow[mi] + k0) : (hrow[mi] + (k0 - NHID));
      v16u a = load_a(arow, lane);
      acc[mi] = wmma_bf16(a, b, acc[mi]);
    }
#if USE_ASYNC
    wait_async0();
    __syncthreads();
#else
    __syncthreads();                       // all reads of cur done
    if (i + 1 < 64) stage((i + 1) * 32, (i + 1) & 1);
    __syncthreads();
#endif
  }

  int ncol = n0_blk + w * 16 + (lane & 15);
#pragma unroll
  for (int mi = 0; mi < 4; ++mi) {
#pragma unroll
    for (int r = 0; r < 8; ++r) {
      int mm = mi * 16 + ((lane < 16) ? r : r + 8);
      gbuf[(size_t)mm * G4H + ncol] = acc[mi][r];
    }
  }
}

// gates: i,f,g,o ; c' = sig(f)*c + sig(i)*tanh(g); h = sig(o)*tanh(c')
__global__ void k_gates(const float* __restrict__ gbuf,
                        const float* __restrict__ bc,
                        float* __restrict__ cst, float* __restrict__ hst,
                        uint16_t* __restrict__ hbf,
                        uint16_t* __restrict__ seqout, int t) {
  int idx = blockIdx.x * 256 + threadIdx.x;         // 64*1024
  int b = idx >> 10;
  int ch = idx & 1023;
  const float* g = gbuf + (size_t)b * G4H;
  float gi = g[ch] + bc[ch];
  float gf = g[NHID + ch] + bc[NHID + ch];
  float gg = g[2 * NHID + ch] + bc[2 * NHID + ch];
  float go = g[3 * NHID + ch] + bc[3 * NHID + ch];
  float cn = sigm(gf) * cst[idx] + sigm(gi) * tanhf(gg);
  float hn = sigm(go) * tanhf(cn);
  cst[idx] = cn;
  hst[idx] = hn;
  uint16_t hb = f2bf(hn);
  hbf[idx] = hb;
  seqout[(size_t)t * BATCH * NHID + idx] = hb;
}

// ---------------- top logits: [8192,1024] @ topWT^T -> [8192,224] ----------------
__global__ void k_top_gemm(const uint16_t* __restrict__ H,
                           const uint16_t* __restrict__ topwt,
                           float* __restrict__ toplog) {
  int lane = threadIdx.x & 31;
  int gid = blockIdx.x * 8 + (threadIdx.x >> 5);    // 7168 waves = 512 x 14
  int nt = gid % 14, mt = gid / 14;
  int m0 = mt * 16, n0 = nt * 16;
  const uint16_t* arow = H + (size_t)(m0 + (lane & 15)) * NHID;
  v8f acc = zero8f();
#pragma unroll 4
  for (int k0 = 0; k0 < NHID; k0 += 32) {
    v16u a = load_a(arow + k0, lane);
    v16u b = load_b_nk(topwt, NHID, n0, k0, lane);
    acc = wmma_bf16(a, b, acc);
  }
#pragma unroll
  for (int r = 0; r < 8; ++r) {
    int mm = m0 + ((lane < 16) ? r : r + 8);
    toplog[(size_t)mm * NCLASSES + n0 + (lane & 15)] = acc[r];
  }
}

// ---------------- class bucketing ----------------
__global__ void k_zero_cnt(int* __restrict__ cnt) {
  int i = threadIdx.x;
  if (i < NCLASSES) cnt[i] = 0;
}
__global__ void k_scatter(const int* __restrict__ targets,
                          int* __restrict__ cnt, int* __restrict__ rowlist) {
  int n = blockIdx.x * 256 + threadIdx.x;           // 8192
  int c = targets[n] / NPC;
  int slot = atomicAdd(&cnt[c], 1);
  rowlist[c * NROWS + slot] = n;
}

// ---------------- bottom logits: class-grouped gathered-A WMMA GEMM ----------------
__global__ void k_bot_gemm(const uint16_t* __restrict__ H,
                           const float* __restrict__ botW,
                           const int* __restrict__ rowlist,
                           const int* __restrict__ cnt,
                           float* __restrict__ botlog) {
  int cls = blockIdx.x / 15;
  int nt = blockIdx.x % 15;
  int n0 = nt * 16;
  int cc = cnt[cls];
  if (cc == 0) return;                               // uniform per block
  int lane = threadIdx.x;
  int nn = n0 + (lane & 15);
  const float* Wc = botW + (size_t)cls * NHID * NPC;
  const int* rl = rowlist + cls * NROWS;
  int kb = (lane < 16) ? 0 : 16;
  for (int m0 = 0; m0 < cc; m0 += 16) {
    int slot = m0 + (lane & 15);
    int row = (slot < cc) ? rl[slot] : -1;
    const uint16_t* arow = (row >= 0) ? (H + (size_t)row * NHID) : H;
    v8f acc = zero8f();
    for (int k0 = 0; k0 < NHID; k0 += 32) {
      if (k0 + 32 < NHID && nn < NPC)
        __builtin_prefetch(&Wc[(size_t)(k0 + 32 + kb) * NPC + nn], 0, 1);
      v16u a = zero16u();
      if (row >= 0) a = load_a(arow + k0, lane);
      v16u b;
#pragma unroll
      for (int j = 0; j < 16; ++j) {
        float w = (nn < NPC) ? Wc[(size_t)(k0 + kb + j) * NPC + nn] : 0.f;
        b[j] = f2bf(w);
      }
      acc = wmma_bf16(a, b, acc);
    }
#pragma unroll
    for (int r = 0; r < 8; ++r) {
      int s2 = m0 + ((lane < 16) ? r : r + 8);
      if (s2 < cc && nn < NPC) {
        int r2 = rl[s2];
        botlog[(size_t)r2 * NPC_PAD + nn] = acc[r];
      }
    }
  }
}

// ---------------- softmax(top)*softmax(bot) pick -> target_probs ----------------
__global__ void k_combine(const float* __restrict__ toplog,
                          const float* __restrict__ botlog,
                          const int* __restrict__ targets,
                          const float* __restrict__ topb,
                          const float* __restrict__ botb,
                          float* __restrict__ out) {
  int lane = threadIdx.x & 31;
  int row = blockIdx.x * 8 + (threadIdx.x >> 5);    // 8192 rows
  int tgt = targets[row];
  int c = tgt / NPC;
  int pb = tgt % NPC;

  const float* tl = toplog + (size_t)row * NCLASSES;
  float tv[7];
  float mx = -1e30f;
#pragma unroll
  for (int j = 0; j < 7; ++j) {
    int idx = lane + j * 32;
    tv[j] = tl[idx] + topb[idx];
    mx = fmaxf(mx, tv[j]);
  }
  mx = wave_max(mx);
  float s = 0.f;
#pragma unroll
  for (int j = 0; j < 7; ++j) s += __expf(tv[j] - mx);
  s = wave_sum(s);
  float ptop = __expf(tl[c] + topb[c] - mx) / s;

  const float* bl = botlog + (size_t)row * NPC_PAD;
  const float* bb = botb + (size_t)c * NPC;
  float bv[8];
  float mxb = -1e30f;
#pragma unroll
  for (int j = 0; j < 8; ++j) {
    int idx = lane + j * 32;
    bv[j] = (idx < NPC) ? (bl[idx] + bb[idx]) : -1e30f;
    mxb = fmaxf(mxb, bv[j]);
  }
  mxb = wave_max(mxb);
  float sb = 0.f;
#pragma unroll
  for (int j = 0; j < 8; ++j) {
    int idx = lane + j * 32;
    if (idx < NPC) sb += __expf(bv[j] - mxb);
  }
  sb = wave_sum(sb);
  float pbot = __expf(bl[pb] + bb[pb] - mxb) / sb;

  if (lane == 0) out[row] = ptop * pbot;
}

__global__ void k_write_states(const float* __restrict__ hst,
                               const float* __restrict__ cst,
                               float* __restrict__ out) {
  int i = blockIdx.x * 256 + threadIdx.x;           // 2*64*1024 = 131072
  out[NROWS + i] = hst[i];
  out[NROWS + 131072 + i] = cst[i];
}

// ---------------- launcher ----------------
extern "C" void kernel_launch(void* const* d_in, const int* in_sizes, int n_in,
                              void* d_out, int out_size, void* d_ws, size_t ws_size,
                              hipStream_t stream) {
  (void)in_sizes; (void)n_in; (void)out_size; (void)ws_size;
  const int*   input_ids = (const int*)d_in[0];
  const int*   targets   = (const int*)d_in[1];
  const float* h0        = (const float*)d_in[2];
  const float* c0        = (const float*)d_in[3];
  const float* emb_W     = (const float*)d_in[4];
  const float* W_ih      = (const float*)d_in[5];
  const float* W_hh      = (const float*)d_in[6];
  const float* b_ih      = (const float*)d_in[7];
  const float* b_hh      = (const float*)d_in[8];
  const float* top_W     = (const float*)d_in[9];
  const float* top_b     = (const float*)d_in[10];
  const float* bot_W     = (const float*)d_in[11];
  const float* bot_b     = (const float*)d_in[12];
  float* out = (float*)d_out;

  char* p = (char*)d_ws;
  auto carve = [&](size_t bytes) -> char* {
    char* r = p;
    p += (bytes + 255) & ~(size_t)255;
    return r;
  };
  uint16_t* wcat   = (uint16_t*)carve((size_t)2 * G4H * KCAT * 2);    // 32 MB
  float*    bcomb  = (float*)   carve((size_t)2 * G4H * 4);
  uint16_t* topwt  = (uint16_t*)carve((size_t)NCLASSES * NHID * 2);
  uint16_t* bufA   = (uint16_t*)carve((size_t)NROWS * NHID * 2);      // 16 MB
  uint16_t* bufB   = (uint16_t*)carve((size_t)NROWS * NHID * 2);      // 16 MB
  float*    hst    = (float*)   carve((size_t)2 * BATCH * NHID * 4);
  float*    cst    = (float*)   carve((size_t)2 * BATCH * NHID * 4);
  uint16_t* hbf    = (uint16_t*)carve((size_t)BATCH * NHID * 2);
  float*    gbuf   = (float*)   carve((size_t)BATCH * G4H * 4);
  float*    toplog = (float*)   carve((size_t)NROWS * NCLASSES * 4);  // 7.3 MB
  float*    botlog = (float*)   carve((size_t)NROWS * NPC_PAD * 4);   // 7.9 MB
  int*      cnt    = (int*)     carve((size_t)NCLASSES * 4);
  int*      rowlist= (int*)     carve((size_t)NCLASSES * NROWS * 4);  // 7.3 MB

  k_prep_wcat <<<65536, 256, 0, stream>>>(W_ih, W_hh, wcat);
  k_prep_bias <<<32, 256, 0, stream>>>(b_ih, b_hh, bcomb);
  k_prep_topwt<<<896, 256, 0, stream>>>(top_W, topwt);
  k_embed     <<<32768, 256, 0, stream>>>(input_ids, emb_W, bufA);

  // LSTM: layer 0 reads bufA -> bufB ; layer 1 reads bufB -> bufA
  for (int l = 0; l < 2; ++l) {
    const uint16_t* xseq = (l == 0) ? bufA : bufB;
    uint16_t* seqout     = (l == 0) ? bufB : bufA;
    k_init_state<<<256, 256, 0, stream>>>(h0 + (size_t)l * BATCH * NHID,
                                          c0 + (size_t)l * BATCH * NHID,
                                          hst + (size_t)l * BATCH * NHID,
                                          cst + (size_t)l * BATCH * NHID, hbf);
    const uint16_t* wl = wcat + (size_t)l * G4H * KCAT;
    const float* bl = bcomb + (size_t)l * G4H;
    float* cl = cst + (size_t)l * BATCH * NHID;
    float* hl = hst + (size_t)l * BATCH * NHID;
    for (int t = 0; t < TSEQ; ++t) {
      k_step_gemm<<<32, 256, 0, stream>>>(xseq, hbf, wl, gbuf, t);
      k_gates    <<<256, 256, 0, stream>>>(gbuf, bl, cl, hl, hbf, seqout, t);
    }
  }

  // hierarchical softmax
  k_top_gemm<<<896, 256, 0, stream>>>(bufA, topwt, toplog);
  k_zero_cnt<<<1, 256, 0, stream>>>(cnt);
  k_scatter <<<32, 256, 0, stream>>>(targets, cnt, rowlist);
  k_bot_gemm<<<NCLASSES * 15, 32, 0, stream>>>(bufA, bot_W, rowlist, cnt, botlog);
  k_combine <<<1024, 256, 0, stream>>>(toplog, botlog, targets, top_b, bot_b, out);
  k_write_states<<<512, 256, 0, stream>>>(hst, cst, out);
}
